// MultiHeadAttentionBlock_24223615550046
// MI455X (gfx1250) — matrix operand order repaired and checked by reading, not compiled, not run on hardware
//
#include <hip/hip_runtime.h>

typedef unsigned int u32;
typedef __attribute__((ext_vector_type(16))) __bf16 v16bf;
typedef __attribute__((ext_vector_type(8)))  float  v8f;

#define WMMA_BF16(a, b, c) \
  __builtin_amdgcn_wmma_f32_16x16x32_bf16(false, (a), false, (b), (short)0, (c), false, false)

namespace {
constexpr int kB  = 2;
constexpr int kS  = 4096;
constexpr int kC  = 512;
constexpr int kNH = 8;
constexpr int kHD = 64;
constexpr int kNG = 32;
constexpr int kM  = kB * kS;   // 8192
constexpr int k3C = 3 * kC;    // 1536
}

__device__ __forceinline__ v8f vzero8() {
  v8f z;
#pragma unroll
  for (int i = 0; i < 8; ++i) z[i] = 0.f;
  return z;
}

// ---------------------------------------------------------------------------
// Kernel 1: GroupNorm statistics.  One block per (batch, group).
// ---------------------------------------------------------------------------
__global__ __launch_bounds__(256) void gn_stats_kernel(const float* __restrict__ x,
                                                       float* __restrict__ stats) {
  const int b = blockIdx.x >> 5;
  const int g = blockIdx.x & 31;
  const float* base = x + (size_t)b * kS * kC + g * 16;
  float sum = 0.f, sq = 0.f;
  for (int idx = threadIdx.x; idx < kS * 4; idx += 256) {
    const int s = idx >> 2, q4 = idx & 3;
    const float4 v = *(const float4*)(base + (size_t)s * kC + q4 * 4);
    sum += v.x + v.y + v.z + v.w;
    sq  += v.x * v.x + v.y * v.y + v.z * v.z + v.w * v.w;
  }
#pragma unroll
  for (int off = 16; off > 0; off >>= 1) {
    sum += __shfl_xor(sum, off, 32);
    sq  += __shfl_xor(sq,  off, 32);
  }
  __shared__ float red[2][8];
  const int wave = threadIdx.x >> 5, lane = threadIdx.x & 31;
  if (lane == 0) { red[0][wave] = sum; red[1][wave] = sq; }
  __syncthreads();
  if (threadIdx.x == 0) {
    float ts = 0.f, tq = 0.f;
#pragma unroll
    for (int w = 0; w < 8; ++w) { ts += red[0][w]; tq += red[1][w]; }
    const float inv_n = 1.0f / 65536.0f;
    const float mean = ts * inv_n;
    const float var  = tq * inv_n - mean * mean;
    stats[(blockIdx.x) * 2 + 0] = mean;
    stats[(blockIdx.x) * 2 + 1] = rsqrtf(var + 1e-6f);
  }
}

// ---------------------------------------------------------------------------
// Kernel 2: fused GroupNorm + QKV GEMM.  M=8192, K=512, N=1536.
// ---------------------------------------------------------------------------
__global__ __launch_bounds__(256) void qkv_gemm_kernel(
    const float* __restrict__ x, const float* __restrict__ stats,
    const float* __restrict__ gscale, const float* __restrict__ gbias,
    const float* __restrict__ w, const float* __restrict__ bias,
    __bf16* __restrict__ qb, __bf16* __restrict__ kb, __bf16* __restrict__ vb) {
  __shared__ __align__(16) __bf16 As[128][34];   // [m][k]
  __shared__ __align__(16) __bf16 Bs[128][34];   // [n][k] (transposed)
  const int tid  = threadIdx.x;
  const int wave = tid >> 5, lane = tid & 31;
  const int half = lane >> 4, l16 = lane & 15;
  const int wm = (wave >> 1) * 32, wn = (wave & 1) * 64;
  const int tm = blockIdx.y * 128;
  const int tn = blockIdx.x * 128;

  v8f acc[2][4];
#pragma unroll
  for (int i = 0; i < 2; ++i)
#pragma unroll
    for (int j = 0; j < 4; ++j) acc[i][j] = vzero8();

  for (int k0 = 0; k0 < kC; k0 += 32) {
    // A tile with fused GroupNorm, float4 loads (4 channels share one group).
#pragma unroll
    for (int i = 0; i < 4; ++i) {
      const int idx = tid + 256 * i;          // 1024 float4 chunks
      const int mm = idx >> 3, q4 = idx & 7;
      const int m = tm + mm, c = k0 + q4 * 4;
      const int bb = m >> 12;
      const float mean = stats[((bb << 5) + (c >> 4)) * 2 + 0];
      const float rstd = stats[((bb << 5) + (c >> 4)) * 2 + 1];
      const float4 v = *(const float4*)(x + (size_t)m * kC + c);
      As[mm][q4 * 4 + 0] = (__bf16)((v.x - mean) * rstd * gscale[c + 0] + gbias[c + 0]);
      As[mm][q4 * 4 + 1] = (__bf16)((v.y - mean) * rstd * gscale[c + 1] + gbias[c + 1]);
      As[mm][q4 * 4 + 2] = (__bf16)((v.z - mean) * rstd * gscale[c + 2] + gbias[c + 2]);
      As[mm][q4 * 4 + 3] = (__bf16)((v.w - mean) * rstd * gscale[c + 3] + gbias[c + 3]);
    }
    // B tile: w[k][n] -> Bs[n][k], float4 over n.
#pragma unroll
    for (int i = 0; i < 4; ++i) {
      const int idx = tid + 256 * i;          // 1024 float4 chunks
      const int kk = idx >> 5, nn4 = idx & 31;
      const float4 v = *(const float4*)(w + (size_t)(k0 + kk) * k3C + tn + nn4 * 4);
      Bs[nn4 * 4 + 0][kk] = (__bf16)v.x;
      Bs[nn4 * 4 + 1][kk] = (__bf16)v.y;
      Bs[nn4 * 4 + 2][kk] = (__bf16)v.z;
      Bs[nn4 * 4 + 3][kk] = (__bf16)v.w;
    }
    __syncthreads();

    v16bf afr[2], bfr[4];
#pragma unroll
    for (int mi = 0; mi < 2; ++mi) {
      const int row = wm + mi * 16 + l16;
#pragma unroll
      for (int e = 0; e < 16; ++e)
        afr[mi][e] = As[row][e + 8 * half + 8 * (e >> 3)];
    }
#pragma unroll
    for (int ni = 0; ni < 4; ++ni) {
      const int col = wn + ni * 16 + l16;
#pragma unroll
      for (int e = 0; e < 16; ++e)
        bfr[ni][e] = Bs[col][e + 8 * half + 8 * (e >> 3)];
    }
#pragma unroll
    for (int mi = 0; mi < 2; ++mi)
#pragma unroll
      for (int ni = 0; ni < 4; ++ni)
        acc[mi][ni] = WMMA_BF16(afr[mi], bfr[ni], acc[mi][ni]);
    __syncthreads();
  }

  // Epilogue: bias, split q/k/v; head/which computed per 16-wide subtile.
#pragma unroll
  for (int ni = 0; ni < 4; ++ni) {
    const int n_base = tn + wn + ni * 16;
    const int head = n_base / 192;
    const int wb = n_base % 192;
    const int which = wb >> 6;
    const int d = (wb & 63) + l16;
    const int n = n_base + l16;
    const float bv = bias[n];
#pragma unroll
    for (int mi = 0; mi < 2; ++mi)
#pragma unroll
      for (int r = 0; r < 8; ++r) {
        const int m = tm + wm + mi * 16 + r + 8 * half;
        const float v = acc[mi][ni][r] + bv;
        const int bb = m >> 12, ss = m & (kS - 1);
        const size_t dst = (((size_t)(bb * kNH + head)) * kS + ss) * kHD + d;
        if (which == 0)      qb[dst] = (__bf16)(v * 0.125f);   // fold 1/sqrt(hd)
        else if (which == 1) kb[dst] = (__bf16)v;
        else                 vb[dst] = (__bf16)v;
      }
  }
}

// ---------------------------------------------------------------------------
// Kernel 3: flash attention.  K/V tiles staged with async global->LDS B128
// copies (ASYNCcnt), Q fragments pinned in registers, online softmax.
// ---------------------------------------------------------------------------
__global__ __launch_bounds__(256) void attn_kernel(
    const __bf16* __restrict__ qb, const __bf16* __restrict__ kb,
    const __bf16* __restrict__ vb, __bf16* __restrict__ hb) {
  __shared__ __align__(16) __bf16 Ks[128][72];   // 144B row stride (16B aligned)
  __shared__ __align__(16) __bf16 Vs[128][72];
  __shared__ __align__(16) __bf16 Ps[8][16][68]; // per-wave P staging (16x64)
  const int tid  = threadIdx.x;
  const int wave = tid >> 5, lane = tid & 31;
  const int half = lane >> 4, l16 = lane & 15;
  const int bh = blockIdx.y;
  const int bb = bh >> 3, head = bh & 7;
  const int q0 = blockIdx.x * 128;
  const __bf16* qp = qb + ((size_t)bh * kS + q0) * kHD;
  const __bf16* kp = kb + (size_t)bh * kS * kHD;
  const __bf16* vp = vb + (size_t)bh * kS * kHD;

  const u32 ldsK = (u32)(size_t)&Ks[0][0];
  const u32 ldsV = (u32)(size_t)&Vs[0][0];

  // Q fragments in A layout stay in registers for the whole kernel.
  v16bf qfr[2];
#pragma unroll
  for (int ks = 0; ks < 2; ++ks)
#pragma unroll
    for (int e = 0; e < 16; ++e)
      qfr[ks][e] = qp[(size_t)(16 * wave + l16) * kHD + ks * 32 + e + 8 * half + 8 * (e >> 3)];

  float mrow[8], lrow[8];
  v8f oacc[4];
#pragma unroll
  for (int r = 0; r < 8; ++r) { mrow[r] = -3.0e38f; lrow[r] = 0.f; }
#pragma unroll
  for (int ni = 0; ni < 4; ++ni) oacc[ni] = vzero8();

  for (int kv0 = 0; kv0 < kS; kv0 += 128) {
    // Async copy K,V 128x64 bf16 tiles to LDS: 1024 16B chunks each.
    const __bf16* kbase = kp + (size_t)kv0 * kHD;
    const __bf16* vbase = vp + (size_t)kv0 * kHD;
#pragma unroll
    for (int i = 0; i < 4; ++i) {
      const int idx = tid + 256 * i;
      const u32 goff = (u32)idx * 16u;                         // global byte offset
      const u32 loff = (u32)(idx >> 3) * 144u + (u32)(idx & 7) * 16u;
      const u32 lk = ldsK + loff;
      const u32 lv = ldsV + loff;
      asm volatile("global_load_async_to_lds_b128 %0, %1, %2"
                   :: "v"(lk), "v"(goff), "s"(kbase) : "memory");
      asm volatile("global_load_async_to_lds_b128 %0, %1, %2"
                   :: "v"(lv), "v"(goff), "s"(vbase) : "memory");
    }
    asm volatile("s_wait_asynccnt 0x0" ::: "memory");
    __syncthreads();

    // S = Q * K^T  (q pre-scaled).
    v8f sacc[8];
#pragma unroll
    for (int t = 0; t < 8; ++t) sacc[t] = vzero8();
#pragma unroll
    for (int ks = 0; ks < 2; ++ks)
#pragma unroll
      for (int t = 0; t < 8; ++t) {
        v16bf bfr;
#pragma unroll
        for (int e = 0; e < 16; ++e)
          bfr[e] = Ks[t * 16 + l16][ks * 32 + e + 8 * half + 8 * (e >> 3)];
        sacc[t] = WMMA_BF16(qfr[ks], bfr, sacc[t]);
      }

    // Online softmax update.  Row (r + 8*half) lives in the lane's half.
    float alpha[8];
#pragma unroll
    for (int r = 0; r < 8; ++r) {
      float mx = -3.0e38f;
#pragma unroll
      for (int t = 0; t < 8; ++t) mx = fmaxf(mx, sacc[t][r]);
#pragma unroll
      for (int off = 1; off < 16; off <<= 1) mx = fmaxf(mx, __shfl_xor(mx, off, 32));
      const float mn = fmaxf(mrow[r], mx);
      alpha[r] = __expf(mrow[r] - mn);
      mrow[r] = mn;
      float rs = 0.f;
#pragma unroll
      for (int t = 0; t < 8; ++t) {
        const float p = __expf(sacc[t][r] - mn);
        sacc[t][r] = p;
        rs += p;
      }
#pragma unroll
      for (int off = 1; off < 16; off <<= 1) rs += __shfl_xor(rs, off, 32);
      lrow[r] = lrow[r] * alpha[r] + rs;
    }
#pragma unroll
    for (int ni = 0; ni < 4; ++ni)
#pragma unroll
      for (int r = 0; r < 8; ++r) oacc[ni][r] *= alpha[r];

    // O += P @ V, two 64-key half passes through per-wave LDS staging.
#pragma unroll
    for (int hc = 0; hc < 2; ++hc) {
#pragma unroll
      for (int t = 0; t < 4; ++t)
#pragma unroll
        for (int r = 0; r < 8; ++r)
          Ps[wave][r + 8 * half][t * 16 + l16] = (__bf16)sacc[hc * 4 + t][r];
#pragma unroll
      for (int kcl = 0; kcl < 2; ++kcl) {
        const int kc = hc * 2 + kcl;
        v16bf a;
#pragma unroll
        for (int e = 0; e < 16; ++e)
          a[e] = Ps[wave][l16][kcl * 32 + e + 8 * half + 8 * (e >> 3)];
#pragma unroll
        for (int ni = 0; ni < 4; ++ni) {
          v16bf bfr;
#pragma unroll
          for (int e = 0; e < 16; ++e)
            bfr[e] = Vs[kc * 32 + e + 8 * half + 8 * (e >> 3)][ni * 16 + l16];
          oacc[ni] = WMMA_BF16(a, bfr, oacc[ni]);
        }
      }
    }
    __syncthreads();
  }

  // Finalize: O /= l, write h as bf16 [B][S][C].
#pragma unroll
  for (int r = 0; r < 8; ++r) {
    const float inv = 1.0f / lrow[r];
    const int srow = q0 + 16 * wave + r + 8 * half;
#pragma unroll
    for (int ni = 0; ni < 4; ++ni) {
      const int d = ni * 16 + l16;
      hb[((size_t)(bb * kS + srow)) * kC + head * kHD + d] = (__bf16)(oacc[ni][r] * inv);
    }
  }
}

// ---------------------------------------------------------------------------
// Kernel 4: output projection + bias + residual.  M=8192, K=512, N=512.
// ---------------------------------------------------------------------------
__global__ __launch_bounds__(256) void proj_gemm_kernel(
    const __bf16* __restrict__ hb, const float* __restrict__ w,
    const float* __restrict__ bias, const float* __restrict__ x,
    float* __restrict__ out) {
  __shared__ __align__(16) __bf16 As[128][40];   // 80B row stride (16B aligned)
  __shared__ __align__(16) __bf16 Bs[128][34];
  const int tid  = threadIdx.x;
  const int wave = tid >> 5, lane = tid & 31;
  const int half = lane >> 4, l16 = lane & 15;
  const int wm = (wave >> 1) * 32, wn = (wave & 1) * 64;
  const int tm = blockIdx.y * 128;
  const int tn = blockIdx.x * 128;

  v8f acc[2][4];
#pragma unroll
  for (int i = 0; i < 2; ++i)
#pragma unroll
    for (int j = 0; j < 4; ++j) acc[i][j] = vzero8();

  for (int k0 = 0; k0 < kC; k0 += 32) {
    // A tile: h already bf16, 16B vector loads/stores.
#pragma unroll
    for (int i = 0; i < 2; ++i) {
      const int idx = tid + 256 * i;         // 512 16B chunks
      const int mm = idx >> 2, ch = idx & 3;
      const uint4 v = ((const uint4*)(hb + (size_t)(tm + mm) * kC + k0))[ch];
      *(uint4*)&As[mm][ch * 8] = v;
    }
    // B tile: w_proj[k][n] -> Bs[n][k], float4 over n.
#pragma unroll
    for (int i = 0; i < 4; ++i) {
      const int idx = tid + 256 * i;         // 1024 float4 chunks
      const int kk = idx >> 5, nn4 = idx & 31;
      const float4 v = *(const float4*)(w + (size_t)(k0 + kk) * kC + tn + nn4 * 4);
      Bs[nn4 * 4 + 0][kk] = (__bf16)v.x;
      Bs[nn4 * 4 + 1][kk] = (__bf16)v.y;
      Bs[nn4 * 4 + 2][kk] = (__bf16)v.z;
      Bs[nn4 * 4 + 3][kk] = (__bf16)v.w;
    }
    __syncthreads();

    v16bf afr[2], bfr[4];
#pragma unroll
    for (int mi = 0; mi < 2; ++mi) {
      const int row = wm + mi * 16 + l16;
#pragma unroll
      for (int e = 0; e < 16; ++e)
        afr[mi][e] = As[row][e + 8 * half + 8 * (e >> 3)];
    }
#pragma unroll
    for (int ni = 0; ni < 4; ++ni) {
      const int col = wn + ni * 16 + l16;
#pragma unroll
      for (int e = 0; e < 16; ++e)
        bfr[ni][e] = Bs[col][e + 8 * half + 8 * (e >> 3)];
    }
#pragma unroll
    for (int mi = 0; mi < 2; ++mi)
#pragma unroll
      for (int ni = 0; ni < 4; ++ni)
        acc[mi][ni] = WMMA_BF16(afr[mi], bfr[ni], acc[mi][ni]);
    __syncthreads();
  }

  // Epilogue: + b_proj + residual x, f32 output.
#pragma unroll
  for (int mi = 0; mi < 2; ++mi)
#pragma unroll
    for (int ni = 0; ni < 4; ++ni)
#pragma unroll
      for (int r = 0; r < 8; ++r) {
        const int m = tm + wm + mi * 16 + r + 8 * half;
        const int n = tn + wn + ni * 16 + l16;
        out[(size_t)m * kC + n] = acc[mi][ni][r] + bias[n] + x[(size_t)m * kC + n];
      }
}

// ---------------------------------------------------------------------------
// Launch
// ---------------------------------------------------------------------------
extern "C" void kernel_launch(void* const* d_in, const int* in_sizes, int n_in,
                              void* d_out, int out_size, void* d_ws, size_t ws_size,
                              hipStream_t stream) {
  (void)in_sizes; (void)n_in; (void)out_size; (void)ws_size;
  const float* x      = (const float*)d_in[0];
  const float* gscale = (const float*)d_in[1];
  const float* gbias  = (const float*)d_in[2];
  const float* wqkv   = (const float*)d_in[3];
  const float* bqkv   = (const float*)d_in[4];
  const float* wproj  = (const float*)d_in[5];
  const float* bproj  = (const float*)d_in[6];
  float* out = (float*)d_out;

  char* ws = (char*)d_ws;
  float* stats = (float*)ws;                                   // 64*2 f32
  const size_t qkv_elems = (size_t)kB * kNH * kS * kHD;        // 4Mi
  __bf16* qbuf = (__bf16*)(ws + 1024);
  __bf16* kbuf = qbuf + qkv_elems;
  __bf16* vbuf = kbuf + qkv_elems;
  __bf16* hbuf = vbuf + qkv_elems;                             // kB*kS*kC == qkv_elems

  gn_stats_kernel<<<dim3(kB * kNG), dim3(256), 0, stream>>>(x, stats);
  qkv_gemm_kernel<<<dim3(k3C / 128, kM / 128), dim3(256), 0, stream>>>(
      x, stats, gscale, gbias, wqkv, bqkv, qbuf, kbuf, vbuf);
  attn_kernel<<<dim3(kS / 128, kB * kNH), dim3(256), 0, stream>>>(
      qbuf, kbuf, vbuf, hbuf);
  proj_gemm_kernel<<<dim3(kC / 128, kM / 128), dim3(256), 0, stream>>>(
      hbuf, wproj, bproj, x, out);
}
